// WindowAttention_58617713655993
// MI455X (gfx1250) — matrix operand order repaired and checked by reading, not compiled, not run on hardware
//
#include <hip/hip_runtime.h>
#include <stdint.h>

// ---------------------------------------------------------------------------
// CDNA5 (gfx1250) Swin window attention, bf16 WMMA pipeline.
//   kernels: repack_w, bias_combine, gemm_qkv, win_attn, gemm_proj
// Workspace requirement: ~423 MB (see offsets in kernel_launch).
// ---------------------------------------------------------------------------

typedef __bf16 v16bf __attribute__((ext_vector_type(16)));
typedef float  v8f   __attribute__((ext_vector_type(8)));

struct FragU { uint4 lo, hi; };

static __device__ __forceinline__ v16bf frag_cast(uint4 lo, uint4 hi) {
  FragU f{lo, hi};
  return __builtin_bit_cast(v16bf, f);
}

// fp32 -> bf16 with round-to-nearest-even
static __device__ __forceinline__ uint32_t f2bf1(float f) {
  uint32_t u = __builtin_bit_cast(uint32_t, f);
  return (u + 0x7FFFu + ((u >> 16) & 1u)) >> 16;
}
static __device__ __forceinline__ uint32_t pack2bf(float a, float b) {
  return f2bf1(a) | (f2bf1(b) << 16);
}

static __device__ __forceinline__ v8f wmma_bf16(v16bf a, v16bf b, v8f c) {
  // D = A(16x32) * B(32x16) + C, f32 accumulate
  return __builtin_amdgcn_wmma_f32_16x16x32_bf16(false, a, false, b,
                                                 (short)0, c, false, false);
}

// ---------------------------------------------------------------------------
// Repack a fp32 weight W[K][Nn] (row-major) into bf16 B-fragment layout:
// fragment (nt, kt) stored at dword ((nt*NKT + kt)*256 + lane*8 + v),
// lane holds column n = nt*16 + lane%16, dword v = K pair
// (kt*32 + (v/4)*16 + (lane/16)*8 + 2*(v%4), +1)  -- per ISA 7.12.2 B-matrix layout.
// ---------------------------------------------------------------------------
__global__ void repack_w(const float* __restrict__ W, uint32_t* __restrict__ out,
                         int K, int Nn) {
  int idx = blockIdx.x * 256 + threadIdx.x;
  int total = (K * Nn) >> 1;
  if (idx >= total) return;
  int v    = idx & 7;
  int lane = (idx >> 3) & 31;
  int frag = idx >> 8;
  int NKT  = K >> 5;
  int kt   = frag % NKT;
  int nt   = frag / NKT;
  int n    = nt * 16 + (lane & 15);
  int k0   = kt * 32 + (v >> 2) * 16 + (lane >> 4) * 8 + 2 * (v & 3);
  out[idx] = pack2bf(W[(size_t)k0 * Nn + n], W[(size_t)(k0 + 1) * Nn + n]);
}

// bm[((w*16+h)*49+i)*49+j] = rel_table[rel_index[i*49+j]][h] + mask[w][i][j]
__global__ void bias_combine(const float* __restrict__ mask,
                             const float* __restrict__ rel_table,
                             const int* __restrict__ rel_index,
                             float* __restrict__ bm) {
  const int NN = 49 * 49;           // 2401
  int idx = blockIdx.x * 256 + threadIdx.x;
  if (idx >= 64 * 16 * NN) return;
  int ij = idx % NN;
  int wh = idx / NN;
  int h  = wh % 16;
  int w  = wh / 16;
  bm[idx] = rel_table[rel_index[ij] * 16 + h] + mask[(size_t)w * NN + ij];
}

// ---------------------------------------------------------------------------
// QKV GEMM: [100352 x 512] fp32 @ repacked [512 x 1536] -> bf16 qkv [M][1536]
// Block = 256 thr (8 waves), tile M=64 x N=512 (grid.y selects 512-col section)
// A is converted fp32->bf16 into LDS (row stride 520 bf16 to rotate banks).
// ---------------------------------------------------------------------------
__global__ __launch_bounds__(256) void gemm_qkv(
    const float* __restrict__ x, const uint32_t* __restrict__ wB,
    const float* __restrict__ qkv_b, uint16_t* __restrict__ qkvB) {
  __shared__ uint32_t As[64 * 260];   // 64 rows x 520 bf16 (260 dwords)
  const int tid = threadIdx.x;
  const int m0  = blockIdx.x * 64;
  const int gz  = blockIdx.y;

  for (int t = tid; t < 64 * 256; t += 256) {
    int row = t >> 8, cp = t & 255;
    const float2 f =
        *reinterpret_cast<const float2*>(x + (size_t)(m0 + row) * 512 + 2 * cp);
    As[row * 260 + cp] = pack2bf(f.x, f.y);
  }
  __syncthreads();

  const int wave = tid >> 5, lane = tid & 31;
  const int half = lane >> 4, lm = lane & 15;
  const int gcol0 = gz * 512 + wave * 64;

  v8f acc[4][4] = {};
  for (int kt = 0; kt < 16; ++kt) {
    v16bf A[4], B[4];
#pragma unroll
    for (int mt = 0; mt < 4; ++mt) {
      int dw = (mt * 16 + lm) * 260 + kt * 16 + half * 4;
      A[mt] = frag_cast(*reinterpret_cast<const uint4*>(&As[dw]),
                        *reinterpret_cast<const uint4*>(&As[dw + 8]));
    }
#pragma unroll
    for (int nt = 0; nt < 4; ++nt) {
      size_t base = ((size_t)(((gcol0 >> 4) + nt) * 16 + kt)) * 256 + lane * 8;
      B[nt] = frag_cast(*reinterpret_cast<const uint4*>(wB + base),
                        *reinterpret_cast<const uint4*>(wB + base + 4));
    }
#pragma unroll
    for (int mt = 0; mt < 4; ++mt)
#pragma unroll
      for (int nt = 0; nt < 4; ++nt)
        acc[mt][nt] = wmma_bf16(A[mt], B[nt], acc[mt][nt]);
  }

#pragma unroll
  for (int nt = 0; nt < 4; ++nt) {
    int c = gcol0 + nt * 16 + lm;
    float bias = qkv_b[c];
#pragma unroll
    for (int mt = 0; mt < 4; ++mt) {
      int i0 = m0 + mt * 16 + half * 8;
#pragma unroll
      for (int r = 0; r < 8; ++r)
        qkvB[(size_t)(i0 + r) * 1536 + c] = (uint16_t)f2bf1(acc[mt][nt][r] + bias);
    }
  }
}

// ---------------------------------------------------------------------------
// Fused window attention: one wave per (window b, head h).
//   S(64x64 pad) = q(49x32) k^T  -> +bias+mask, softmax -> P -> LDS bf16
//   O(49x32)     = P v   (v transposed into zero-padded LDS)
// ---------------------------------------------------------------------------
__global__ __launch_bounds__(32) void win_attn(
    const uint16_t* __restrict__ qkvB, const float* __restrict__ bm,
    uint16_t* __restrict__ attnO) {
  __shared__ uint16_t Psh[64 * 72];   // P, row stride 72 bf16 (144B, /16 ok)
  __shared__ uint16_t vT[32 * 72];    // v^T [d][j], zero padded j in [49,64)
  const int b = blockIdx.x, h = blockIdx.y;
  const int lane = threadIdx.x, half = lane >> 4, lm = lane & 15;
  const size_t qbase = (size_t)b * 49 * 1536;

  uint32_t* vT32 = reinterpret_cast<uint32_t*>(vT);
  for (int i = lane; i < 32 * 36; i += 32) vT32[i] = 0u;
  __syncthreads();
  for (int t = lane; t < 49 * 16; t += 32) {   // stage v^T
    int j = t >> 4, dp = t & 15;
    uint32_t val = *reinterpret_cast<const uint32_t*>(
        qkvB + qbase + (size_t)j * 1536 + 1024 + h * 32 + dp * 2);
    vT[(2 * dp) * 72 + j]     = (uint16_t)(val & 0xffffu);
    vT[(2 * dp + 1) * 72 + j] = (uint16_t)(val >> 16);
  }

  // k^T B-fragments: lane = token column, contiguous head-dim in memory
  v16bf Bk[4];
#pragma unroll
  for (int nt = 0; nt < 4; ++nt) {
    int m = nt * 16 + lm;
    uint4 lo = {0, 0, 0, 0}, hi = {0, 0, 0, 0};
    if (m < 49) {
      const uint4* p = reinterpret_cast<const uint4*>(
          qkvB + qbase + (size_t)m * 1536 + 512 + h * 32 + half * 16);
      lo = p[0]; hi = p[1];
    }
    Bk[nt] = frag_cast(lo, hi);
  }

  const float* bmp = bm + (size_t)((b & 63) * 16 + h) * 2401;

  for (int mt = 0; mt < 4; ++mt) {
    int tq = mt * 16 + lm;
    uint4 lo = {0, 0, 0, 0}, hi = {0, 0, 0, 0};
    if (tq < 49) {
      const uint16_t* p = qkvB + qbase + (size_t)tq * 1536 + h * 32;
      lo = *reinterpret_cast<const uint4*>(p + half * 16);
      hi = *reinterpret_cast<const uint4*>(p + half * 16 + 8);
    }
    v16bf Aq = frag_cast(lo, hi);

    v8f st[4] = {};
#pragma unroll
    for (int nt = 0; nt < 4; ++nt) st[nt] = wmma_bf16(Aq, Bk[nt], st[nt]);

#pragma unroll
    for (int r = 0; r < 8; ++r) {
      int i = mt * 16 + half * 8 + r;
      float sv[4];
#pragma unroll
      for (int nt = 0; nt < 4; ++nt) {
        int j = nt * 16 + lm;
        float s = st[nt][r] * 0.1767766952966369f;  // 1/sqrt(32)
        if (i < 49 && j < 49) s += bmp[i * 49 + j];
        if (j >= 49) s = -1e30f;                    // pad columns
        sv[nt] = s;
      }
      float mx = fmaxf(fmaxf(sv[0], sv[1]), fmaxf(sv[2], sv[3]));
#pragma unroll
      for (int off = 1; off < 16; off <<= 1) mx = fmaxf(mx, __shfl_xor(mx, off, 32));
      float e[4], sum = 0.f;
#pragma unroll
      for (int nt = 0; nt < 4; ++nt) { e[nt] = __expf(sv[nt] - mx); sum += e[nt]; }
#pragma unroll
      for (int off = 1; off < 16; off <<= 1) sum += __shfl_xor(sum, off, 32);
      float inv = 1.0f / sum;
#pragma unroll
      for (int nt = 0; nt < 4; ++nt)
        Psh[i * 72 + nt * 16 + lm] = (uint16_t)f2bf1(e[nt] * inv);
    }
  }
  __syncthreads();

  // v B-fragments from vT (contiguous j per lane column d)
  v16bf Bv[2][2];
#pragma unroll
  for (int kt2 = 0; kt2 < 2; ++kt2)
#pragma unroll
    for (int nd = 0; nd < 2; ++nd) {
      const uint16_t* p = &vT[(nd * 16 + lm) * 72 + kt2 * 32 + half * 16];
      Bv[kt2][nd] = frag_cast(*reinterpret_cast<const uint4*>(p),
                              *reinterpret_cast<const uint4*>(p + 8));
    }

#pragma unroll
  for (int mt = 0; mt < 4; ++mt) {
    v8f oacc[2] = {};
#pragma unroll
    for (int kt2 = 0; kt2 < 2; ++kt2) {
      const uint16_t* p = &Psh[(mt * 16 + lm) * 72 + kt2 * 32];
      v16bf Ap = frag_cast(*reinterpret_cast<const uint4*>(p + half * 16),
                           *reinterpret_cast<const uint4*>(p + half * 16 + 8));
#pragma unroll
      for (int nd = 0; nd < 2; ++nd)
        oacc[nd] = wmma_bf16(Ap, Bv[kt2][nd], oacc[nd]);
    }
#pragma unroll
    for (int nd = 0; nd < 2; ++nd)
#pragma unroll
      for (int r = 0; r < 8; ++r) {
        int i = mt * 16 + half * 8 + r;
        if (i < 49)
          attnO[((size_t)b * 49 + i) * 512 + h * 32 + nd * 16 + lm] =
              (uint16_t)f2bf1(oacc[nd][r]);
      }
  }
}

// ---------------------------------------------------------------------------
// Projection GEMM: bf16 [100352 x 512] @ repacked [512 x 512] + b -> fp32 out
// ---------------------------------------------------------------------------
__global__ __launch_bounds__(256) void gemm_proj(
    const uint16_t* __restrict__ aO, const uint32_t* __restrict__ wB,
    const float* __restrict__ proj_b, float* __restrict__ out) {
  __shared__ uint32_t As[64 * 260];
  const int tid = threadIdx.x;
  const int m0  = blockIdx.x * 64;

  for (int t = tid; t < 64 * 64; t += 256) {
    int row = t >> 6, u4 = t & 63;
    uint4 v = *reinterpret_cast<const uint4*>(aO + (size_t)(m0 + row) * 512 + u4 * 8);
    *reinterpret_cast<uint4*>(&As[row * 260 + u4 * 4]) = v;
  }
  __syncthreads();

  const int wave = tid >> 5, lane = tid & 31;
  const int half = lane >> 4, lm = lane & 15;
  const int gcol0 = wave * 64;

  v8f acc[4][4] = {};
  for (int kt = 0; kt < 16; ++kt) {
    v16bf A[4], B[4];
#pragma unroll
    for (int mt = 0; mt < 4; ++mt) {
      int dw = (mt * 16 + lm) * 260 + kt * 16 + half * 4;
      A[mt] = frag_cast(*reinterpret_cast<const uint4*>(&As[dw]),
                        *reinterpret_cast<const uint4*>(&As[dw + 8]));
    }
#pragma unroll
    for (int nt = 0; nt < 4; ++nt) {
      size_t base = ((size_t)(((gcol0 >> 4) + nt) * 16 + kt)) * 256 + lane * 8;
      B[nt] = frag_cast(*reinterpret_cast<const uint4*>(wB + base),
                        *reinterpret_cast<const uint4*>(wB + base + 4));
    }
#pragma unroll
    for (int mt = 0; mt < 4; ++mt)
#pragma unroll
      for (int nt = 0; nt < 4; ++nt)
        acc[mt][nt] = wmma_bf16(A[mt], B[nt], acc[mt][nt]);
  }

#pragma unroll
  for (int nt = 0; nt < 4; ++nt) {
    int c = gcol0 + nt * 16 + lm;
    float bias = proj_b[c];
#pragma unroll
    for (int mt = 0; mt < 4; ++mt) {
      int i0 = m0 + mt * 16 + half * 8;
#pragma unroll
      for (int r = 0; r < 8; ++r)
        out[(size_t)(i0 + r) * 512 + c] = acc[mt][nt][r] + bias;
    }
  }
}

// ---------------------------------------------------------------------------
extern "C" void kernel_launch(void* const* d_in, const int* in_sizes, int n_in,
                              void* d_out, int out_size, void* d_ws, size_t ws_size,
                              hipStream_t stream) {
  (void)in_sizes; (void)n_in; (void)out_size; (void)ws_size;
  const float* x      = (const float*)d_in[0];
  const float* mask   = (const float*)d_in[1];
  const float* qkv_w  = (const float*)d_in[2];
  const float* qkv_b  = (const float*)d_in[3];
  const float* proj_w = (const float*)d_in[4];
  const float* proj_b = (const float*)d_in[5];
  const float* rel_t  = (const float*)d_in[6];
  const int*   rel_i  = (const int*)d_in[7];

  char* ws = (char*)d_ws;
  // workspace layout (bytes), all 16B aligned:
  uint16_t* qkvB   = (uint16_t*)(ws);               // 100352*1536*2 = 308,281,344
  uint16_t* attnO  = (uint16_t*)(ws + 308281344);   // 100352*512*2  = 102,760,448
  uint32_t* qkvWB  = (uint32_t*)(ws + 411041792);   // 512*1536/2*4  =   1,572,864
  uint32_t* projWB = (uint32_t*)(ws + 412614656);   // 512*512/2*4   =     524,288
  float*    bmbuf  = (float*)   (ws + 413138944);   // 64*16*2401*4  =   9,834,496

  repack_w<<<(512 * 1536 / 2 + 255) / 256, 256, 0, stream>>>(qkv_w, qkvWB, 512, 1536);
  repack_w<<<(512 * 512 / 2 + 255) / 256, 256, 0, stream>>>(proj_w, projWB, 512, 512);
  bias_combine<<<(64 * 16 * 2401 + 255) / 256, 256, 0, stream>>>(mask, rel_t, rel_i, bmbuf);
  gemm_qkv<<<dim3(1568, 3), 256, 0, stream>>>(x, qkvWB, qkv_b, qkvB);
  win_attn<<<dim3(2048, 16), 32, 0, stream>>>(qkvB, bmbuf, attnO);
  gemm_proj<<<1568, 256, 0, stream>>>(attnO, projWB, proj_b, (float*)d_out);
}
